// VAE_12111807774821
// MI455X (gfx1250) — matrix operand order repaired and checked by reading, not compiled, not run on hardware
//
#include <hip/hip_runtime.h>
#include <hip/hip_bf16.h>

typedef __attribute__((ext_vector_type(16))) _Float16 v16h;
typedef __attribute__((ext_vector_type(8)))  float    v8f;
typedef __attribute__((ext_vector_type(4)))  unsigned int u32x4;
typedef __attribute__((ext_vector_type(8)))  int      i32x8;
typedef __attribute__((ext_vector_type(4)))  int      i32x4;

#define BATCH 64
#define PTS   1024
#define NN    16      // top-k size (incl. self)
#define NBR   15      // neighbors used (excl. self)
#define NMOM  32      // nmoments
#define HID   128
#define LAT   2

// ---------------------------------------------------------------------------
// Tensor Data Mover: 1-D f32 tile, global -> LDS. Issued once per workgroup
// (wave 0). D# built per CDNA5 ISA 8.3/8.4.
// ---------------------------------------------------------------------------
__device__ __forceinline__ void tdm_load_f32(void* ldsDst, const void* gsrc,
                                             unsigned int n /* f32 elems, tile_dim0<=65535 */)
{
    unsigned long long ga = (unsigned long long)(uintptr_t)gsrc;
    unsigned int lds = (unsigned int)(uintptr_t)ldsDst;   // LDS addr = low 32 bits of flat
    u32x4 g0;
    g0[0] = 1u;                                           // count=1, is_restore=0, no gather
    g0[1] = lds;                                          // lds_addr
    g0[2] = (unsigned int)ga;                             // global_addr[31:0]
    g0[3] = (unsigned int)(ga >> 32) | 0x80000000u;       // global_addr[56:32] | type=2<<30
    i32x8 g1;
    g1[0] = 0x00020000;                                   // data_size = 2 (4 bytes)
    g1[1] = (int)((n & 0xFFFFu) << 16);                   // tensor_dim0[15:0]
    g1[2] = (int)((n >> 16) | (1u << 16));                // tensor_dim0[31:16] | tensor_dim1=1
    g1[3] = (int)((n & 0xFFFFu) << 16);                   // tile_dim0 = n
    g1[4] = 1;                                            // tile_dim1=1, tile_dim2=0
    g1[5] = (int)n;                                       // tensor_dim0_stride[31:0]
    g1[6] = 0;
    g1[7] = 0;
    i32x4 gz = {0, 0, 0, 0};
#if __clang_major__ >= 23
    i32x8 gz8 = {0, 0, 0, 0, 0, 0, 0, 0};
    __builtin_amdgcn_tensor_load_to_lds(g0, g1, gz, gz, gz8, 0);
#else
    __builtin_amdgcn_tensor_load_to_lds(g0, g1, gz, gz, 0);
#endif
}

// wave32 half-swap reduce: ds_swizzle SWAPX16 (xor=0x10, and=0x1f)
__device__ __forceinline__ float swap16_add(float v)
{
    int o = __builtin_amdgcn_ds_swizzle(__float_as_int(v), 0x401f);
    return v + __int_as_float(o);
}

// ---------------------------------------------------------------------------
// prep: zterm[b][c] = bias_x[c] + bias_z[c] + (z @ Wz)[b][c]   (64 x 64, padded)
//       wf16[k][c]  = f16(W_meas_x | W_vect_x) padded to 64x64
// ---------------------------------------------------------------------------
__global__ void vae_prep(const float* __restrict__ Wmx, const float* __restrict__ bmx,
                         const float* __restrict__ Wmz, const float* __restrict__ bmz,
                         const float* __restrict__ Wvx, const float* __restrict__ bvx,
                         const float* __restrict__ Wvz, const float* __restrict__ bvz,
                         const float* __restrict__ zsrc, int zin, int d, int dout,
                         float* __restrict__ zterm, _Float16* __restrict__ wf16)
{
    int tid = blockIdx.x * blockDim.x + threadIdx.x;
    int stride = gridDim.x * blockDim.x;
    int ncols = dout + NMOM;
    int twod = 2 * d;

    for (int t = tid; t < BATCH * 64; t += stride) {
        int b = t >> 6, c = t & 63;
        float v = 0.0f;
        if (c < dout) {
            v = bmx[c] + bmz[c];
            for (int k = 0; k < zin; ++k)
                v = fmaf(zsrc[b * NMOM + k], Wmz[k * dout + c], v);
        } else if (c < ncols) {
            int cc = c - dout;
            v = bvx[cc] + bvz[cc];
            for (int k = 0; k < zin; ++k)
                v = fmaf(zsrc[b * NMOM + k], Wvz[k * NMOM + cc], v);
        }
        zterm[t] = v;
    }
    for (int t = tid; t < 64 * 64; t += stride) {
        int k = t >> 6, c = t & 63;
        float v = 0.0f;
        if (k < twod) {
            if (c < dout)       v = Wmx[k * dout + c];
            else if (c < ncols) v = Wvx[k * NMOM + (c - dout)];
        }
        wf16[t] = (_Float16)v;
    }
}

// ---------------------------------------------------------------------------
// knn: fused distance + top-16. Cloud staged in LDS by the Tensor Data Mover;
// never materializes the (B,P,P) distance tensor (268MB > 192MB L2).
// ---------------------------------------------------------------------------
template <int D>
__global__ void vae_knn(const float* __restrict__ xin, int* __restrict__ idx)
{
    extern __shared__ float pts[];
    int b = blockIdx.y;
    const float* xb = xin + (size_t)b * PTS * D;
    if (threadIdx.x < 32) {
        tdm_load_f32(pts, xb, PTS * D);
        __builtin_amdgcn_s_wait_tensorcnt(0);
    }
    __syncthreads();

    int p = blockIdx.x * blockDim.x + threadIdx.x;
    float xp[D];
#pragma unroll
    for (int k = 0; k < D; ++k) xp[k] = pts[p * D + k];

    float bd[NN];
    int   bi[NN];
#pragma unroll
    for (int j = 0; j < NN; ++j) { bd[j] = 3.4e38f; bi[j] = 0; }

    for (int q = 0; q < PTS; ++q) {
        float d2 = 0.0f;
#pragma unroll
        for (int k = 0; k < D; ++k) {
            float df = xp[k] - pts[q * D + k];
            d2 = fmaf(df, df, d2);
        }
        if (d2 < bd[NN - 1]) {
            bd[NN - 1] = d2; bi[NN - 1] = q;
#pragma unroll
            for (int j = NN - 1; j > 0; --j) {
                if (bd[j] < bd[j - 1]) {
                    float td = bd[j]; bd[j] = bd[j - 1]; bd[j - 1] = td;
                    int   ti = bi[j]; bi[j] = bi[j - 1]; bi[j - 1] = ti;
                } else break;
            }
        }
    }
    int* op = idx + ((size_t)b * PTS + p) * NN;
#pragma unroll
    for (int j = 0; j < NN; ++j) op[j] = bi[j];
}

// ---------------------------------------------------------------------------
// block GEMM, point-aligned tiles: tile == point p; rows 0..14 = its 15
// neighbor pairs, row 15 = pad (masked). K padded to 64, N padded to 64.
// Cloud staged by TDM; A-fragments gathered branch-free from LDS;
// 8x v_wmma_f32_16x16x32_f16 per tile; epilogue: bias+zterm+relu, x-mean =
// in-register column sum + ds_swizzle half-swap + plain store (no atomics),
// z-mean = register accumulation + one atomic per (lane, ntile).
// ---------------------------------------------------------------------------
template <int D>
__global__ void __launch_bounds__(128)
vae_block_gemm(const float* __restrict__ xin, const int* __restrict__ idx,
               const _Float16* __restrict__ wf16, const float* __restrict__ zterm,
               int dout, float* __restrict__ out_x, float* __restrict__ out_z)
{
    extern __shared__ char smem[];
    _Float16* wl  = (_Float16*)smem;                 // 64x64 f16 weights (8KB)
    float*    pad = (float*)(smem + 64 * 64 * 2);    // 64 zeroed floats (OOB target)
    float*    xl  = pad + 64;                        // PTS*D f32 cloud

    int b = blockIdx.y;
    int tid = threadIdx.x;

    if (tid < 32) tdm_load_f32(xl, xin + (size_t)b * PTS * D, PTS * D);  // async DMA
    for (int t = tid; t < 64 * 64; t += blockDim.x) wl[t] = wf16[t];     // overlap
    if (tid < 64) pad[tid] = 0.0f;
    __builtin_prefetch(idx + (size_t)b * PTS * NN, 0, 3);                // global_prefetch_b8
    if (tid < 32) __builtin_amdgcn_s_wait_tensorcnt(0);
    __syncthreads();

    int wave = tid >> 5, lane = tid & 31;
    int laneHi = (lane >= 16) ? 1 : 0;
    int r = lane & 15;
    int ncols = dout + NMOM;

    float zacc[4] = {0.f, 0.f, 0.f, 0.f};

    // B fragments (reused by all 4 tiles this wave processes)
    v16h bf[2][4];
#pragma unroll
    for (int ks = 0; ks < 2; ++ks)
#pragma unroll
        for (int nt = 0; nt < 4; ++nt)
#pragma unroll
            for (int j = 0; j < 16; ++j) {
                int K = ((j < 8) ? j : j + 8) + (laneHi ? 8 : 0) + ks * 32;
                bf[ks][nt][j] = wl[K * 64 + nt * 16 + r];
            }

    for (int it = 0; it < 4; ++it) {
        int p = (blockIdx.x * 4 + wave) * 4 + it;         // tile == point, 0..1023
        // A-row r holds pair (center p, neighbor r); row 15 is pad (masked below)
        int n1 = (r < NBR) ? (r + 1) : NBR;               // clamp pad row to a valid idx
        int nbr = idx[((size_t)b * PTS + p) * NN + n1];
        int baseC = 64 + p * D;                           // offsets relative to `pad`
        int baseN = 64 + nbr * D;

        v16h a[2];
#pragma unroll
        for (int f = 0; f < 2; ++f)
#pragma unroll
            for (int j = 0; j < 16; ++j) {
                int K = ((j < 8) ? j : j + 8) + (laneHi ? 8 : 0) + f * 32;
                // branch-free pad: OOB K -> zeroed pad zone; always one ds_load_b32
                int off = (K < D) ? (baseC + K)
                                  : ((K < 2 * D) ? (baseN + (K - D)) : j);
                a[f][j] = (_Float16)pad[off];
            }

        v8f acc[4] = {};
#pragma unroll
        for (int ks = 0; ks < 2; ++ks)
#pragma unroll
            for (int nt = 0; nt < 4; ++nt)
                acc[nt] = __builtin_amdgcn_wmma_f32_16x16x32_f16(
                    false, a[ks], false, bf[ks][nt],
                    (short)0, acc[nt], false, false);

#pragma unroll
        for (int nt = 0; nt < 4; ++nt) {
            int col = nt * 16 + r;                        // fixed per (lane, nt)
            float zt = zterm[b * 64 + col];
            float colsum = 0.0f;
#pragma unroll
            for (int rr = 0; rr < 8; ++rr) {
                float v = acc[nt][rr] + zt;
                v = (v > 0.0f) ? v : 0.0f;
                if (rr == 7 && laneHi) v = 0.0f;          // mask pad row 15
                colsum += v;
            }
            // fold the two half-wave row groups: rows 0-7 (+) rows 8-14
            float tot = swap16_add(colsum);
            if (col < dout) {
                if (!laneHi)
                    out_x[((size_t)b * PTS + p) * dout + col] = tot * (1.0f / (float)NBR);
            } else if (col < ncols) {
                zacc[nt] += colsum;                       // keep per-lane; folded by atomics
            }
        }
    }
#pragma unroll
    for (int nt = 0; nt < 4; ++nt) {
        int col = nt * 16 + r;
        if (col >= dout && col < ncols)
            atomicAdd(out_z + b * NMOM + (col - dout),
                      zacc[nt] * (1.0f / ((float)NBR * (float)PTS)));
    }
}

// ---------------------------------------------------------------------------
// head: fc1 -> BatchNorm(train) -> relu -> fc21/fc22 -> reparametrize
// ---------------------------------------------------------------------------
__global__ void vae_head(const float* __restrict__ zsrc,
                         const float* __restrict__ W1,  const float* __restrict__ b1,
                         const float* __restrict__ gamma, const float* __restrict__ beta,
                         const float* __restrict__ W21, const float* __restrict__ b21,
                         const float* __restrict__ W22, const float* __restrict__ b22,
                         const float* __restrict__ eps,
                         float* __restrict__ out_mu, float* __restrict__ out_lv,
                         float* __restrict__ zlat)
{
    __shared__ float hs[BATCH * HID];
    int tid = threadIdx.x;
    for (int t = tid; t < BATCH * HID; t += blockDim.x) {
        int b = t >> 7, j = t & (HID - 1);
        float v = b1[j];
        for (int k = 0; k < NMOM; ++k)
            v = fmaf(zsrc[b * NMOM + k], W1[k * HID + j], v);
        hs[t] = v;
    }
    __syncthreads();
    if (tid < HID) {
        int j = tid;
        float mu = 0.0f;
        for (int b = 0; b < BATCH; ++b) mu += hs[b * HID + j];
        mu *= (1.0f / BATCH);
        float var = 0.0f;
        for (int b = 0; b < BATCH; ++b) {
            float dv = hs[b * HID + j] - mu;
            var = fmaf(dv, dv, var);
        }
        var *= (1.0f / BATCH);
        float inv = rsqrtf(var + 1e-5f);
        for (int b = 0; b < BATCH; ++b) {
            float v = (hs[b * HID + j] - mu) * inv * gamma[j] + beta[j];
            hs[b * HID + j] = (v > 0.0f) ? v : 0.0f;
        }
    }
    __syncthreads();
    if (tid < BATCH * LAT) {
        int b = tid >> 1, l = tid & 1;
        float m  = b21[l];
        float lv = b22[l];
        for (int j = 0; j < HID; ++j) {
            float h = hs[b * HID + j];
            m  = fmaf(h, W21[j * LAT + l], m);
            lv = fmaf(h, W22[j * LAT + l], lv);
        }
        out_mu[b * LAT + l] = m;
        out_lv[b * LAT + l] = lv;
        zlat[b * NMOM + l] = eps[b * LAT + l] * __expf(0.5f * lv) + m;
    }
}

// ---------------------------------------------------------------------------
// host orchestration
// ---------------------------------------------------------------------------
struct EBp { const float *Wmx, *bmx, *Wmz, *bmz, *Wvx, *bvx, *Wvz, *bvz; };

static inline EBp eb_from(void* const* d_in, int base) {
    EBp e;
    e.Wmx = (const float*)d_in[base + 0]; e.bmx = (const float*)d_in[base + 1];
    e.Wmz = (const float*)d_in[base + 2]; e.bmz = (const float*)d_in[base + 3];
    e.Wvx = (const float*)d_in[base + 4]; e.bvx = (const float*)d_in[base + 5];
    e.Wvz = (const float*)d_in[base + 6]; e.bvz = (const float*)d_in[base + 7];
    return e;
}

static void run_block(int D, int dout, int zin,
                      const float* xin, float* xout,
                      const float* zsrc, float* zout,
                      const EBp& e, float* zterm, _Float16* wf16, int* idx,
                      hipStream_t stream)
{
    hipMemsetAsync(zout, 0, (size_t)BATCH * NMOM * sizeof(float), stream);

    vae_prep<<<dim3(16), 256, 0, stream>>>(e.Wmx, e.bmx, e.Wmz, e.bmz,
                                           e.Wvx, e.bvx, e.Wvz, e.bvz,
                                           zsrc, zin, D, dout, zterm, wf16);

    size_t lds_knn = (size_t)PTS * D * sizeof(float);
    size_t lds_g   = 64 * 64 * 2 + 64 * sizeof(float) + (size_t)PTS * D * sizeof(float);
    dim3 gk(PTS / 256, BATCH);
    dim3 gg(64, BATCH);                       // 64 wg * 4 waves * 4 tiles = 1024 points

    if (D == 2) {
        vae_knn<2><<<gk, 256, lds_knn, stream>>>(xin, idx);
        vae_block_gemm<2><<<gg, 128, lds_g, stream>>>(xin, idx, wf16, zterm, dout, xout, zout);
    } else if (D == 15) {
        vae_knn<15><<<gk, 256, lds_knn, stream>>>(xin, idx);
        vae_block_gemm<15><<<gg, 128, lds_g, stream>>>(xin, idx, wf16, zterm, dout, xout, zout);
    } else {
        vae_knn<30><<<gk, 256, lds_knn, stream>>>(xin, idx);
        vae_block_gemm<30><<<gg, 128, lds_g, stream>>>(xin, idx, wf16, zterm, dout, xout, zout);
    }
}

extern "C" void kernel_launch(void* const* d_in, const int* in_sizes, int n_in,
                              void* d_out, int out_size, void* d_ws, size_t ws_size,
                              hipStream_t stream)
{
    (void)in_sizes; (void)n_in; (void)out_size; (void)ws_size;

    // setup_inputs() order: x, eps, x_dec, then params pytree (sorted keys):
    // bn_beta, bn_gamma, eb1..eb6 (meas_x.W,b, meas_z.W,b, vect_x.W,b, vect_z.W,b),
    // fc1.W,b, fc21.W,b, fc22.W,b
    const float* x    = (const float*)d_in[0];
    const float* eps  = (const float*)d_in[1];
    const float* xdec = (const float*)d_in[2];
    const float* bn_beta  = (const float*)d_in[3];
    const float* bn_gamma = (const float*)d_in[4];
    EBp eb1 = eb_from(d_in, 5),  eb2 = eb_from(d_in, 13), eb3 = eb_from(d_in, 21);
    EBp eb4 = eb_from(d_in, 29), eb5 = eb_from(d_in, 37), eb6 = eb_from(d_in, 45);
    const float* fc1W  = (const float*)d_in[53];
    const float* fc1b  = (const float*)d_in[54];
    const float* fc21W = (const float*)d_in[55];
    const float* fc21b = (const float*)d_in[56];
    const float* fc22W = (const float*)d_in[57];
    const float* fc22b = (const float*)d_in[58];

    // workspace layout (~19MB)
    char* ws = (char*)d_ws;
    const size_t XBUF = (size_t)BATCH * PTS * 30 * sizeof(float);   // 7.86MB
    float* xA   = (float*)ws;
    float* xB   = (float*)(ws + XBUF);
    int*   idx  = (int*)(ws + 2 * XBUF);
    float* zA   = (float*)(ws + 2 * XBUF + (size_t)BATCH * PTS * NN * sizeof(int));
    float* zB   = zA + BATCH * NMOM;
    float* zterm = zB + BATCH * NMOM;                // 64x64
    _Float16* wf16 = (_Float16*)(zterm + 64 * 64);   // 64x64 f16

    float* out     = (float*)d_out;
    float* out_xd  = out;                              // B*P*2
    float* out_mu  = out + (size_t)BATCH * PTS * 2;    // 128
    float* out_lv  = out_mu + BATCH * LAT;             // 128

    // encoder: z0 = 0
    hipMemsetAsync(zA, 0, (size_t)BATCH * NMOM * sizeof(float), stream);
    run_block(2, 15, 1,  x,   xA, zA, zB, eb1, zterm, wf16, idx, stream);
    run_block(15, 30, 32, xA,  xB, zB, zA, eb2, zterm, wf16, idx, stream);
    run_block(30, 1, 32,  xB,  xA, zA, zB, eb3, zterm, wf16, idx, stream);

    // head: fc1 + BN + relu + fc21/22 + reparam; writes mu/logvar + zl (-> zA cols 0..1)
    vae_head<<<1, 256, 0, stream>>>(zB, fc1W, fc1b, bn_gamma, bn_beta,
                                    fc21W, fc21b, fc22W, fc22b, eps,
                                    out_mu, out_lv, zA);

    // decoder
    run_block(2, 15, 2,  xdec, xB, zA, zB, eb4, zterm, wf16, idx, stream);
    run_block(15, 30, 32, xB,  xA, zB, zA, eb5, zterm, wf16, idx, stream);
    run_block(30, 2, 32,  xA, out_xd, zA, zB, eb6, zterm, wf16, idx, stream);
}
